// ExCorrAEVComputer_77653008712446
// MI455X (gfx1250) — matrix operand order repaired and checked by reading, not compiled, not run on hardware
//
#include <hip/hip_runtime.h>
#include <hip/hip_bf16.h>

// ---------------------------------------------------------------------------
// AEV computer for MI455X (gfx1250, wave32).
//
// Output (256, 24, 384) f32 = 9.4MB -> ~0.4us at 23.3TB/s: not HBM bound.
// Work is ~1.7M angular triples x transcendentals -> VALU/latency bound.
// The angular one-hot einsum ('mijkz,mjkp->mipz') is a real matmul
// out[10x32] += A[10xK] * B[Kx32] over K = neighbor pairs, executed with
// v_wmma_f32_16x16x32_f16 (f16 inputs in [0,2], f32 accumulation), keeping
// the 10x32 accumulator in VGPRs (two v8f frags).
//
// Round-3 changes: incremental one-hot A staging (2 u16 stores/chunk instead
// of 16 cmp+sel+stores over persistently-zeroed LDS), and a branch-light
// store epilogue (rows {0,1,8,9} unconditional + one half-wave guard).
// ---------------------------------------------------------------------------

typedef __attribute__((ext_vector_type(16))) _Float16 v16h;
typedef __attribute__((ext_vector_type(8)))  _Float16 v8h;
typedef __attribute__((ext_vector_type(8)))  float    v8f;

#define ATOMS   24
#define PAIR2   276       // C(24,2) pairs (j<k); center overlap killed by fc=0
#define PAIRPAD 288       // 9 chunks * 32 lanes
#define CHUNKS  9
#define NSPEC   4
#define RADSUB  16
#define ANGSUB  32
#define NPP     10        // species-pair classes
#define FEAT    384       // 4*16 + 10*32
#define NWAVES  8
#define BLOCK   256

__constant__ float c_cosZ[8] = { 0.98078528f, 0.83146961f, 0.55557023f, 0.19509032f,
                                -0.19509032f,-0.55557023f,-0.83146961f,-0.98078528f };
__constant__ float c_sinZ[8] = { 0.19509032f, 0.55557023f, 0.83146961f, 0.98078528f,
                                 0.98078528f, 0.83146961f, 0.55557023f, 0.19509032f };
__constant__ float c_shfA[4] = { 0.9f, 1.55f, 2.2f, 2.85f };

__global__ __launch_bounds__(BLOCK)
void aev_kernel(const int* __restrict__ species,
                const float* __restrict__ coords,
                float* __restrict__ out) {
  __shared__ float  sX[ATOMS], sY[ATOMS], sZ[ATOMS];
  __shared__ int    sS[ATOMS];
  __shared__ float4 sGeo[ATOMS][ATOMS];          // (ux, uy, uz, d) -> ds_load_b128
  __shared__ float  sFr[ATOMS][ATOMS];           // fc at rc=5.2 (radial)
  __shared__ float  sFa[ATOMS][ATOMS];           // fc at rc=3.5 (angular)
  __shared__ unsigned int sPair[PAIRPAD];        // j | k<<8 | pair_class<<16
  // K-major transposed staging (per wave): fragment slices are contiguous.
  __shared__ __align__(64) _Float16 sBT[NWAVES][32][32];  // [z][pair q]
  __shared__ __align__(64) _Float16 sAT[NWAVES][16][32];  // [p][pair q]

  const int m    = blockIdx.x;
  const int tid  = threadIdx.x;
  const int lane = tid & 31;
  const int wave = tid >> 5;

  const float* cm = coords + (size_t)m * ATOMS * 3;
  __builtin_prefetch(cm, 0, 0);                  // global_prefetch_b8

  if (tid < ATOMS) {
    sX[tid] = cm[tid * 3 + 0];
    sY[tid] = cm[tid * 3 + 1];
    sZ[tid] = cm[tid * 3 + 2];
    sS[tid] = species[m * ATOMS + tid];
  }
  __syncthreads();

  // pairwise geometry tables
  for (int idx = tid; idx < ATOMS * ATOMS; idx += BLOCK) {
    int i = idx / ATOMS, j = idx - i * ATOMS;
    float dx = sX[j] - sX[i], dy = sY[j] - sY[i], dz = sZ[j] - sZ[i];
    float d2 = dx * dx + dy * dy + dz * dz;
    float d  = (i == j) ? 1.0f : sqrtf(d2);          // ref sets diag dist = 1
    float fr = (i != j && d <= 5.2f) ? (0.5f * __cosf(0.60415243f * d) + 0.5f) : 0.0f;
    float fa = (i != j && d <= 3.5f) ? (0.5f * __cosf(0.89759790f * d) + 0.5f) : 0.0f;
    float inv = 1.0f / d;
    sGeo[i][j] = make_float4(dx * inv, dy * inv, dz * inv, d);
    sFr[i][j] = fr;
    sFa[i][j] = fa;
  }
  // packed pair table: (j<k) decode + species pair class, padded with (0,0)
  for (int idx = tid; idx < PAIRPAD; idx += BLOCK) {
    int j = 0, k = 0;
    if (idx < PAIR2) {
      int off = 0;
      while (off + (ATOMS - 1 - j) <= idx) { off += ATOMS - 1 - j; ++j; }
      k = j + 1 + (idx - off);
    }
    int sj = sS[j], sk = sS[k];
    int lo = sj < sk ? sj : sk, hi = sj < sk ? sk : sj;
    int pcls = lo * NSPEC - (lo * (lo - 1)) / 2 + (hi - lo);  // triu index
    sPair[idx] = (unsigned)j | ((unsigned)k << 8) | ((unsigned)pcls << 16);
  }
  __syncthreads();

  float* om = out + (size_t)m * ATOMS * FEAT;

  // ---------------- radial: 384 (i,r) items, 4 species accumulators --------
  for (int idx = tid; idx < ATOMS * RADSUB; idx += BLOCK) {
    int i = idx >> 4, r = idx & 15;
    float shf = 0.9f + 0.26875f * (float)r;
    float a0 = 0.f, a1 = 0.f, a2 = 0.f, a3 = 0.f;
#pragma unroll
    for (int j = 0; j < ATOMS; ++j) {
      float t = sGeo[i][j].w - shf;
      float v = 0.25f * __expf(-16.0f * t * t) * sFr[i][j];   // 0 for j==i / d>rc
      int s = sS[j];
      a0 += (s == 0) ? v : 0.f;  a1 += (s == 1) ? v : 0.f;
      a2 += (s == 2) ? v : 0.f;  a3 += (s == 3) ? v : 0.f;
    }
    float* oi = om + i * FEAT;
    oi[r] = a0; oi[16 + r] = a1; oi[32 + r] = a2; oi[48 + r] = a3;
  }

  // ---------------- angular via WMMA: one wave per center atom i -----------
  // one-time zero of this wave's A staging; kept zero except one live slot
  {
    unsigned int* az = (unsigned int*)&sAT[wave][0][0];   // 256 dwords
#pragma unroll
    for (int t = 0; t < 8; ++t) az[lane + 32 * t] = 0u;
  }
  int prev = 0;   // last written one-hot row (already zero)

  for (int i = wave; i < ATOMS; i += NWAVES) {
    v8f acc0 = {};   // features z = 0..15
    v8f acc1 = {};   // features z = 16..31
    for (int c = 0; c < CHUNKS; ++c) {
      const int q = c * 32 + lane;

      // ---- branchless per-pair math (padded entries -> finite, w2 = 0) ----
      unsigned pr = sPair[q];
      int j    = pr & 255;
      int k    = (pr >> 8) & 255;
      int pcls = (pr >> 16) & 255;
      float fj = sFa[i][j], fk = sFa[i][k];
      float w2 = (q < PAIR2) ? 2.0f * fj * fk : 0.0f;
      float4 gj = sGeo[i][j];
      float4 gk = sGeo[i][k];
      float ca = 0.95f * (gj.x * gk.x + gj.y * gk.y + gj.z * gk.z);
      float sa = sqrtf(fmaxf(1.0f - ca * ca, 0.0f));          // sin(arccos(ca))
      float davg = 0.5f * (gj.w + gk.w);
      float f2v[4], f1v[8];
#pragma unroll
      for (int a = 0; a < 4; ++a) {
        float t = davg - c_shfA[a];
        f2v[a] = __expf(-8.0f * t * t);                       // <=1, always finite
      }
#pragma unroll
      for (int b = 0; b < 8; ++b) {
        float cd = ca * c_cosZ[b] + sa * c_sinZ[b];           // cos(theta - z_b)
        float x  = 0.5f * (1.0f + cd);
        float x2 = x * x, x4 = x2 * x2, x8 = x4 * x4, x16 = x8 * x8;
        f1v[b] = x16 * x16;                                   // ^32
      }

      // ---- stage: B row q (K-major transposed), incremental one-hot A -----
#pragma unroll
      for (int a = 0; a < 4; ++a)
#pragma unroll
        for (int b = 0; b < 8; ++b)
          sBT[wave][a * 8 + b][lane] = (_Float16)(f2v[a] * f1v[b]);
      sAT[wave][prev][lane] = (_Float16)0.0f;   // clear stale slot (in-order DS)
      sAT[wave][pcls][lane] = (_Float16)w2;     // set new slot
      prev = pcls;

      __asm__ volatile("s_wait_dscnt 0" ::: "memory");        // wave-local RAW

      // ---- aligned vector fragment loads (contiguous K order in A and B: slot s of half h is K = 16h + s) -------------
      const int row = lane & 15;                 // A-matrix M index
      const int col = lane & 15;                 // B-matrix N index
      const int kb  = (lane & 16) ? 16 : 0;      // B half-wave K base
      const int ka  = (lane & 16) ? 16 : 0;      // A half-wave K base
      v16h bf0 = *(const v16h*)&sBT[wave][col][kb];        // 2x ds_load_b128
      v16h bf1 = *(const v16h*)&sBT[wave][col + 16][kb];
      v8h  alo = *(const v8h*)&sAT[wave][row][ka];         // slots 0..7: K=ka..ka+7
      v8h  ahi = *(const v8h*)&sAT[wave][row][ka + 8];     // slots 8..15: K=ka+8..ka+15
      v16h af  = __builtin_shufflevector(alo, ahi,
                   0, 1, 2, 3, 4, 5, 6, 7, 8, 9, 10, 11, 12, 13, 14, 15);

      acc0 = __builtin_amdgcn_wmma_f32_16x16x32_f16(false, af, false, bf0,
                                                    (short)0, acc0, false, false);
      acc1 = __builtin_amdgcn_wmma_f32_16x16x32_f16(false, af, false, bf1,
                                                    (short)0, acc1, false, false);
    }

    // D-matrix f32 layout: VGPR v -> row v + 8*(lane>=16), col = lane%16.
    // p in {0,1,8,9} always valid; p in {2..7} valid for lower half-wave only.
    const int rbase = (lane & 16) ? 8 : 0;
    const int col   = lane & 15;
    float* oi = om + i * FEAT + NSPEC * RADSUB;
#pragma unroll
    for (int v = 0; v < 2; ++v) {
      int p = v + rbase;
      oi[p * ANGSUB + col]      = acc0[v];
      oi[p * ANGSUB + 16 + col] = acc1[v];
    }
    if (!(lane & 16)) {
#pragma unroll
      for (int v = 2; v < 8; ++v) {
        oi[v * ANGSUB + col]      = acc0[v];
        oi[v * ANGSUB + 16 + col] = acc1[v];
      }
    }
  }
}

extern "C" void kernel_launch(void* const* d_in, const int* in_sizes, int n_in,
                              void* d_out, int out_size, void* d_ws, size_t ws_size,
                              hipStream_t stream) {
  (void)n_in; (void)d_ws; (void)ws_size; (void)out_size;
  const int*   species = (const int*)d_in[0];
  const float* coords  = (const float*)d_in[1];
  // d_in[2] (coefficients) is unused by the reference output.
  float* out = (float*)d_out;
  const int M = in_sizes[0] / ATOMS;
  if (M <= 0) return;
  aev_kernel<<<dim3(M), dim3(BLOCK), 0, stream>>>(species, coords, out);
}
